// TreeLSTM_37864431681778
// MI455X (gfx1250) — compile-verified, matrix operand order checked by reading, not run on hardware
//
#include <hip/hip_runtime.h>
#include <hip/hip_bf16.h>
#include <stdint.h>

#define H_ 512
#define N_ 16384
#define B_ 4
#define K_ 512   // inner dim of all GEMMs (== H)

typedef __attribute__((ext_vector_type(16))) __bf16 v16bf;
typedef __attribute__((ext_vector_type(8)))  __bf16 v8bf;
typedef __attribute__((ext_vector_type(4)))  __bf16 v4bf;
typedef __attribute__((ext_vector_type(8)))  float  v8f;
typedef __attribute__((ext_vector_type(4)))  float  v4f;

__device__ __forceinline__ float sigmoidf_(float x) {
    return 1.0f / (1.0f + __expf(-x));
}

// ---------------------------------------------------------------------------
// One-time weight repack: Wx4 = [Wioux; Wfx] -> bf16 (4H x H, row-major j,k),
// Wfh -> bf16, b4 = [bioux; bfx] (f32). Uses HW f32->bf16 convert.
// ---------------------------------------------------------------------------
__global__ void pack_weights(const float* __restrict__ Wioux,
                             const float* __restrict__ Wfx,
                             const float* __restrict__ Wfh,
                             const float* __restrict__ bioux,
                             const float* __restrict__ bfx,
                             __bf16* __restrict__ Wx4,
                             __bf16* __restrict__ WfhB,
                             float* __restrict__ b4) {
    int idx = blockIdx.x * blockDim.x + threadIdx.x;
    const int nWx4 = 4 * H_ * K_;
    const int nWfh = H_ * K_;
    if (idx < nWx4) {
        int j = idx / K_, k = idx - j * K_;
        float v = (j < 3 * H_) ? Wioux[j * K_ + k] : Wfx[(j - 3 * H_) * K_ + k];
        Wx4[idx] = (__bf16)v;
    } else if (idx < nWx4 + nWfh) {
        int t = idx - nWx4;
        WfhB[t] = (__bf16)Wfh[t];
    } else if (idx < nWx4 + nWfh + 4 * H_) {
        int j = idx - nWx4 - nWfh;
        b4[j] = (j < 3 * H_) ? bioux[j] : bfx[j - 3 * H_];
    }
}

// ---------------------------------------------------------------------------
// Tiled GEMM: out[M x Nout] = A[M x 512] * Bw[Nout x 512]^T + bias[Nout]
// A is f32 (converted to bf16 while staging in LDS); Bw is pre-packed bf16.
// GATHER==1: logical row r -> (b, node, child) gather from the h buffer,
//            zeros for children outside the tree.
// GUARD==0:  Mrows is an exact multiple of 256 -> no row bounds checks.
// Workgroup: 256 threads = 8 waves; tile 256(M) x 64(N); K step 32 via LDS.
// Each wave owns 32 rows (2 A-frags) x 64 cols (4 B-frags) = 8 WMMAs/K-step.
// Staging is two-phase (issue all loads, then convert+store) so the 8
// global_load_b128 per thread overlap instead of serializing on loadcnt.
// Core op: v_wmma_f32_16x16x32_bf16.
// ---------------------------------------------------------------------------
template <int GATHER, int GUARD>
__global__ void __launch_bounds__(256)
gemm_bf16_wmma(const float* __restrict__ A,
               const __bf16* __restrict__ Bw,
               const float* __restrict__ bias,
               float* __restrict__ out,
               int Mrows, int Nout, int lvlStart, int lvlCount) {
    __shared__ __align__(16) __bf16 As[256 * 32];
    __shared__ __align__(16) __bf16 Bs[64 * 32];

    const int t    = threadIdx.x;
    const int wid  = t >> 5;
    const int lane = t & 31;
    const int sel  = lane >> 4;   // half-wave select (K-group)
    const int ln   = lane & 15;   // row (A) / col (B) within fragment
    const int m0   = blockIdx.x * 256;
    const int n0   = blockIdx.y * 64;

    v8f acc[2][4] = {{}};

    for (int k0 = 0; k0 < K_; k0 += 32) {
        // ---- phase 1: issue all A-tile loads (8 x float4 per thread) ----
        v4f tav[8];
        #pragma unroll
        for (int i = 0; i < 8; ++i) {
            int idx  = t + 256 * i;        // 2048 float4 granules
            int row  = idx >> 3;
            int q    = idx & 7;            // q-th float4 within the row
            int grow = m0 + row;
            v4f v = {};
            if (!GUARD || grow < Mrows) {
                if (!GATHER) {
                    v = *(const v4f*)&A[(size_t)grow * K_ + (k0 + 4 * q)];
                } else {
                    int L2    = lvlCount * 2;
                    int b     = grow / L2;
                    int rem   = grow - b * L2;
                    int node  = lvlStart + (rem >> 1);
                    int child = 2 * node + 1 + (rem & 1);
                    if (child < N_)
                        v = *(const v4f*)&A[((size_t)b * N_ + child) * (size_t)H_ +
                                            (k0 + 4 * q)];
                }
            }
            tav[i] = v;
        }
        // ---- issue B-tile loads (2 x bf16x4 per thread) ----
        v4bf tbv[2];
        #pragma unroll
        for (int i = 0; i < 2; ++i) {
            int idx = t + 256 * i;         // 512 bf16x4 granules
            int col = idx >> 3;
            int q   = idx & 7;
            tbv[i] = *(const v4bf*)&Bw[(size_t)(n0 + col) * K_ + (k0 + 4 * q)];
        }
        // prefetch next K-slab of the (hot, reused) weight matrix
        if (k0 + 32 < K_)
            __builtin_prefetch((const void*)(Bw + (size_t)n0 * K_ + k0 + 32), 0, 1);

        // ---- phase 2: convert + store to LDS ----
        #pragma unroll
        for (int i = 0; i < 8; ++i) {
            int idx = t + 256 * i;
            int row = idx >> 3;
            int q   = idx & 7;
            v4bf p;
            #pragma unroll
            for (int e = 0; e < 4; ++e) p[e] = (__bf16)tav[i][e]; // v_cvt_pk_bf16_f32
            *(v4bf*)&As[row * 32 + 4 * q] = p;
        }
        #pragma unroll
        for (int i = 0; i < 2; ++i) {
            int idx = t + 256 * i;
            int col = idx >> 3;
            int q   = idx & 7;
            *(v4bf*)&Bs[col * 32 + 4 * q] = tbv[i];
        }
        __syncthreads();

        // ---- A fragments (16x32 bf16) per ISA lane layout ----
        // lanes 0-15: K = {0..7, 16..23}; lanes 16-31: K = {8..15, 24..31}
        v16bf af[2];
        #pragma unroll
        for (int g = 0; g < 2; ++g) {
            int arow = wid * 32 + g * 16 + ln;
            const v8bf lo = *(const v8bf*)&As[arow * 32 + sel * 8];
            const v8bf hi = *(const v8bf*)&As[arow * 32 + 16 + sel * 8];
            #pragma unroll
            for (int e = 0; e < 8; ++e) { af[g][e] = lo[e]; af[g][8 + e] = hi[e]; }
        }
        // ---- 4 B fragments (32x16), each reused by both A fragments ----
        #pragma unroll
        for (int f = 0; f < 4; ++f) {
            v16bf bfv;
            const v8bf lo = *(const v8bf*)&Bs[(f * 16 + ln) * 32 + sel * 8];
            const v8bf hi = *(const v8bf*)&Bs[(f * 16 + ln) * 32 + 16 + sel * 8];
            #pragma unroll
            for (int e = 0; e < 8; ++e) { bfv[e] = lo[e]; bfv[8 + e] = hi[e]; }
            #pragma unroll
            for (int g = 0; g < 2; ++g)
                acc[g][f] = __builtin_amdgcn_wmma_f32_16x16x32_bf16(
                    false, af[g], false, bfv, (short)0, acc[g][f], false, false);
        }
        __syncthreads();
    }

    // ---- epilogue: C/D layout => lanes 0-15: M=v, lanes 16-31: M=v+8 ----
    #pragma unroll
    for (int g = 0; g < 2; ++g) {
        const int gm = m0 + wid * 32 + g * 16 + sel * 8;
        #pragma unroll
        for (int f = 0; f < 4; ++f) {
            int gc = n0 + f * 16 + ln;
            float bb = bias[gc];
            #pragma unroll
            for (int v = 0; v < 8; ++v) {
                int gr = gm + v;
                if (!GUARD || gr < Mrows)
                    out[(size_t)gr * Nout + gc] = acc[g][f][v] + bb;
            }
        }
    }
}

// ---------------------------------------------------------------------------
// Per-level elementwise LSTM cell, 4 j-elements per thread (b128 accesses).
// Y holds [i|o|u|fx] (biases already fused). Fpre holds h_child*Wfh^T + bfh.
// ---------------------------------------------------------------------------
__global__ void tree_cell(const float* __restrict__ Y,
                          const float* __restrict__ Fpre,
                          float* __restrict__ Cbuf,
                          float* __restrict__ Hbuf,
                          int lvlStart, int lvlCount) {
    long long idx    = (long long)blockIdx.x * blockDim.x + threadIdx.x;
    long long totalV = (long long)B_ * lvlCount * (H_ / 4);
    if (idx >= totalV) return;
    int jq = (int)(idx % (H_ / 4));
    int j  = 4 * jq;
    long long rem = idx / (H_ / 4);
    int i = (int)(rem % lvlCount);
    int b = (int)(rem / lvlCount);
    int node = lvlStart + i;

    size_t row4 = ((size_t)b * N_ + node) * (size_t)(4 * H_);
    v4f ig = *(const v4f*)&Y[row4 + j];
    v4f og = *(const v4f*)&Y[row4 + H_ + j];
    v4f ug = *(const v4f*)&Y[row4 + 2 * H_ + j];
    v4f fx = *(const v4f*)&Y[row4 + 3 * H_ + j];

    v4f fsum = {};
    #pragma unroll
    for (int k = 0; k < 2; ++k) {
        int child = 2 * node + 1 + k;
        if (child < N_) {
            size_t rF = ((size_t)b * lvlCount + i) * 2 + k;
            v4f fp = *(const v4f*)&Fpre[rF * H_ + j];
            v4f cc = *(const v4f*)&Cbuf[((size_t)b * N_ + child) * (size_t)H_ + j];
            #pragma unroll
            for (int e = 0; e < 4; ++e)
                fsum[e] += sigmoidf_(fx[e] + fp[e]) * cc[e];
        }
    }
    v4f cn, hn;
    #pragma unroll
    for (int e = 0; e < 4; ++e) {
        cn[e] = sigmoidf_(ig[e]) * tanhf(ug[e]) + fsum[e];
        hn[e] = sigmoidf_(og[e]) * tanhf(cn[e]);
    }
    *(v4f*)&Cbuf[((size_t)b * N_ + node) * (size_t)H_ + j] = cn;
    *(v4f*)&Hbuf[((size_t)b * N_ + node) * (size_t)H_ + j] = hn;
}

// ---------------------------------------------------------------------------
extern "C" void kernel_launch(void* const* d_in, const int* in_sizes, int n_in,
                              void* d_out, int out_size, void* d_ws, size_t ws_size,
                              hipStream_t stream) {
    const float* x     = (const float*)d_in[0];
    const float* Wioux = (const float*)d_in[1];
    const float* bioux = (const float*)d_in[2];
    // d_in[3] (Wiouh) and d_in[4] (biouh) are unused by the reference.
    const float* Wfx   = (const float*)d_in[5];
    const float* bfx   = (const float*)d_in[6];
    const float* Wfh   = (const float*)d_in[7];
    const float* bfh   = (const float*)d_in[8];
    float* Hout = (float*)d_out;

    // Workspace layout (all fully re-written each call before being read):
    float* Y    = (float*)d_ws;                       // B*N x 4H  (512 MB)
    float* Cbuf = Y    + (size_t)B_ * N_ * 4 * H_;    // B*N x H   (128 MB)
    float* Fpre = Cbuf + (size_t)B_ * N_ * H_;        // up to B*8192*2 x H (128 MB)
    __bf16* Wx4  = (__bf16*)(Fpre + (size_t)B_ * N_ * H_);
    __bf16* WfhB = Wx4 + (size_t)4 * H_ * K_;
    float* b4 = (float*)(WfhB + (size_t)H_ * K_);

    // 1) pack weights to bf16 + fused bias vector
    int packTotal = 4 * H_ * K_ + H_ * K_ + 4 * H_;
    pack_weights<<<(packTotal + 255) / 256, 256, 0, stream>>>(
        Wioux, Wfx, Wfh, bioux, bfx, Wx4, WfhB, b4);

    // 2) big dependency-free GEMM: Y = x @ [Wioux;Wfx]^T + [bioux;bfx]
    //    M = 65536 is an exact multiple of the 256-row tile -> GUARD=0.
    {
        dim3 g((B_ * N_) / 256, (4 * H_) / 64);
        gemm_bf16_wmma<0, 0><<<g, 256, 0, stream>>>(x, Wx4, b4, Y,
                                                    B_ * N_, 4 * H_, 0, 0);
    }

    // 3) level sweep: leaves -> root
    for (int d = 14; d >= 0; --d) {
        int start = (1 << d) - 1;
        if (start >= N_) continue;
        int end = (1 << (d + 1)) - 1;
        if (end > N_) end = N_;
        int count = end - start;
        int rows  = B_ * count * 2;

        dim3 g((rows + 255) / 256, H_ / 64);
        gemm_bf16_wmma<1, 1><<<g, 256, 0, stream>>>(Hout, WfhB, bfh, Fpre,
                                                    rows, H_, start, count);

        long long totalV = (long long)B_ * count * (H_ / 4);
        tree_cell<<<(unsigned)((totalV + 255) / 256), 256, 0, stream>>>(
            Y, Fpre, Cbuf, Hout, start, count);
    }
}
